// MotionDiscrim_54107997995210
// MI455X (gfx1250) — matrix operand order repaired and checked by reading, not compiled, not run on hardware
//
#include <hip/hip_runtime.h>

typedef __attribute__((ext_vector_type(16))) __bf16 v16bf;
typedef __attribute__((ext_vector_type(8)))  float  v8f;

#define NFRAMES 16
#define NBATCH  16
#define NCH     512
#define NHW     256
#define NDIM    256

// ---- optional gfx1250 async global->LDS path (guarded; falls back cleanly) ----
#if defined(__has_builtin)
#if __has_builtin(__builtin_amdgcn_global_load_async_to_lds_b128) && \
    __has_builtin(__builtin_amdgcn_s_wait_asynccnt)
#define USE_ASYNC_LDS 1
#endif
#endif

#ifdef USE_ASYNC_LDS
typedef int v4i_t __attribute__((vector_size(16)));
typedef __attribute__((address_space(1))) v4i_t gl_v4i;   // global AS int4
typedef __attribute__((address_space(3))) v4i_t lds_v4i;  // LDS AS int4
#endif

__device__ __forceinline__ void copy16_to_lds(void* lds_ptr, const void* gptr) {
#ifdef USE_ASYNC_LDS
  __builtin_amdgcn_global_load_async_to_lds_b128(
      (gl_v4i*)(unsigned long long)gptr,
      (lds_v4i*)(unsigned)(unsigned long long)lds_ptr, 0, 0);
#else
  *(uint4*)lds_ptr = *(const uint4*)gptr;
#endif
}

__device__ __forceinline__ void async_fence_lds() {
#ifdef USE_ASYNC_LDS
  __builtin_amdgcn_s_wait_asynccnt(0);
#endif
}

__device__ __forceinline__ float leaky(float x) { return x >= 0.f ? x : 0.2f * x; }

// ---------------------------------------------------------------------------
// sent = leaky(bn_batch(cond @ lin_w.T + lin_b))     [16, 256]
// ---------------------------------------------------------------------------
__global__ void k_sent(const float* __restrict__ cond, const float* __restrict__ lin_w,
                       const float* __restrict__ lin_b, const float* __restrict__ g,
                       const float* __restrict__ bt, float* __restrict__ sent) {
  const int d = threadIdx.x;
  float acc[NBATCH];
#pragma unroll
  for (int bb = 0; bb < NBATCH; ++bb) acc[bb] = lin_b[d];
  for (int k = 0; k < NDIM; ++k) {
    const float w = lin_w[d * NDIM + k];
#pragma unroll
    for (int bb = 0; bb < NBATCH; ++bb) acc[bb] += cond[bb * NDIM + k] * w;
  }
  float mean = 0.f;
#pragma unroll
  for (int bb = 0; bb < NBATCH; ++bb) mean += acc[bb];
  mean *= (1.f / NBATCH);
  float var = 0.f;
#pragma unroll
  for (int bb = 0; bb < NBATCH; ++bb) { float dlt = acc[bb] - mean; var += dlt * dlt; }
  var *= (1.f / NBATCH);
  const float rstd = rsqrtf(var + 1e-5f);
  const float gd = g[d], bd = bt[d];
#pragma unroll
  for (int bb = 0; bb < NBATCH; ++bb) {
    float y = (acc[bb] - mean) * rstd * gd + bd;
    sent[bb * NDIM + d] = leaky(y);
  }
}

// ---------------------------------------------------------------------------
// bf16 weight copies: wA1 = bf16(conv1_w[512x512]); wA2 = bf16(conv2_w[:, :512])
// ---------------------------------------------------------------------------
__global__ void k_cvt(const float* __restrict__ w1, const float* __restrict__ w2,
                      __bf16* __restrict__ wA1, __bf16* __restrict__ wA2) {
  const int gid = blockIdx.x * 256 + threadIdx.x;      // 0 .. 512*512-1
  wA1[gid] = (__bf16)w1[gid];
  const int o = gid >> 9, c = gid & 511;
  wA2[gid] = (__bf16)w2[o * 768 + c];
}

// ---------------------------------------------------------------------------
// bias2[b,o] = sum_d conv2_w[o, 512+d] * sent[b,d]
// ---------------------------------------------------------------------------
__global__ void k_bias2(const float* __restrict__ w2, const float* __restrict__ sent,
                        float* __restrict__ bias2) {
  const int gid = blockIdx.x * 256 + threadIdx.x;      // 0 .. 16*512-1
  const int b = gid >> 9, o = gid & 511;
  float acc = 0.f;
  const float* wrow = w2 + (size_t)o * 768 + 512;
  const float* srow = sent + b * NDIM;
  for (int d = 0; d < NDIM; ++d) acc += wrow[d] * srow[d];
  bias2[b * NCH + o] = acc;
}

// ---------------------------------------------------------------------------
// Tiled bf16 WMMA GEMM, double-buffered LDS + software pipelining.
//   hout[f,b,o,hw] = sum_c A[o,c] * Bmat[f,b,c,hw] (+bias)
// STAGE 1: Bmat = xbar[f+1]-xbar[f]      (frame diff on the fly)
// STAGE 2: Bmat = leaky(scale*h1+shift)  (BN1+leaky fused into the loader)
// Block = 256 threads (8 waves), 128x128 output tile, K step 32.
// Wave (wm in 0..3, wn in 0..1) owns 32(M) x 64(N): 2x4 WMMA accumulators.
// ---------------------------------------------------------------------------
#define LDSTRIDE 40   // padded bf16 row stride (conflict-free fragment reads)

template <int STAGE>
__global__ void k_gemm(const float* __restrict__ xbar, const __bf16* __restrict__ wA,
                       const __bf16* __restrict__ hin, const float* __restrict__ scl,
                       const float* __restrict__ shf, const float* __restrict__ bias2,
                       __bf16* __restrict__ hout) {
  const int bid = blockIdx.x;                 // F * 4 * 32 = 2048 blocks
  const int f  = bid >> 7;
  const int r  = bid & 127;
  const int mt = r >> 5;                      // 4 M-tiles of 128
  const int nt = r & 31;                      // 32 N-tiles of 128
  const int m0 = mt * 128;
  const int b  = nt >> 1;                     // 128-wide tile never straddles a batch row
  const int hw0 = (nt & 1) * 128;

  const int tid  = threadIdx.x;
  const int lane = tid & 31;
  const int wave = tid >> 5;
  const int wm = wave & 3, wn = wave >> 2;
  const int l15 = lane & 15;
  const int lhi = lane >> 4;                  // 0 or 1

  __shared__ __bf16 lA[2][128 * LDSTRIDE];    // [m][k]
  __shared__ __bf16 lB[2][128 * LDSTRIDE];    // [n][k]   (k-major per column)

  v8f acc[2][4];
#pragma unroll
  for (int mi = 0; mi < 2; ++mi)
#pragma unroll
    for (int ni = 0; ni < 4; ++ni)
#pragma unroll
      for (int e = 0; e < 8; ++e) acc[mi][ni][e] = 0.f;

  // A-tile loader: 2 threads per row, 16 bf16 (32B) each
  const int arow  = tid >> 1;
  const int ahalf = (tid & 1) * 16;
  const __bf16* wrow = wA + (size_t)(m0 + arow) * NCH + ahalf;

  // B-tile loader: thread -> (n column, k half of 16)
  const int nl = tid & 127;
  const int kh = tid >> 7;                    // 0/1
  const size_t fb512 = (size_t)(f * NBATCH + b) * NCH;
  const float*  x0 = xbar + fb512 * NHW + (hw0 + nl);
  const float*  x1 = x0 + (size_t)NBATCH * NCH * NHW;       // next frame
  const __bf16* hb = (STAGE == 2) ? (hin + fb512 * NHW + (hw0 + nl)) : nullptr;
  const float* scf = (STAGE == 2) ? (scl + f * NCH) : nullptr;
  const float* shfr= (STAGE == 2) ? (shf + f * NCH) : nullptr;

  auto load_tile = [&](int k0, int buf) {
    // --- A tile: pure copy -> async global->LDS when available ---
    __bf16* adst = &lA[buf][arow * LDSTRIDE + ahalf];
    copy16_to_lds(adst, wrow + k0);
    copy16_to_lds(adst + 8, wrow + k0 + 8);
    if (k0 + 64 < NCH) __builtin_prefetch(wrow + k0 + 64, 0, 3);
    // --- B tile: computed, transposed [n][k] ---
    __bf16 tmp[16];
    const int cbase = k0 + kh * 16;
#pragma unroll
    for (int j = 0; j < 16; ++j) {
      const int c = cbase + j;
      float v;
      if (STAGE == 1) {
        const size_t idx = (size_t)c * NHW;
        v = x1[idx] - x0[idx];
      } else {
        const float x = (float)hb[(size_t)c * NHW];
        v = leaky(x * scf[c] + shfr[c]);
      }
      tmp[j] = (__bf16)v;
    }
    uint4* dst = (uint4*)&lB[buf][nl * LDSTRIDE + kh * 16];
    dst[0] = ((const uint4*)tmp)[0];
    dst[1] = ((const uint4*)tmp)[1];
  };

  union Frag { v16bf v; uint4 q[2]; };

  // prologue: stage tile 0
  load_tile(0, 0);
  async_fence_lds();
  __syncthreads();

  for (int k0 = 0; k0 < NCH; k0 += 32) {
    const int cur = (k0 >> 5) & 1;
    const int nxt = cur ^ 1;

    // issue next tile's loads; they overlap the WMMAs below
    if (k0 + 32 < NCH) load_tile(k0 + 32, nxt);

    // --- fragments (ISA 16-bit A 16x32 / B 32x16 wave32 layouts) ---
    Frag af[2], bfr[4];
    const int kb = lhi * 8;
#pragma unroll
    for (int mi = 0; mi < 2; ++mi) {
      const int rr = wm * 32 + mi * 16 + l15;
      af[mi].q[0] = *(const uint4*)&lA[cur][rr * LDSTRIDE + kb];
      af[mi].q[1] = *(const uint4*)&lA[cur][rr * LDSTRIDE + 16 + kb];
    }
    const int koff = lhi * 8;
#pragma unroll
    for (int ni = 0; ni < 4; ++ni) {
      const int cB = wn * 64 + ni * 16 + l15;
      bfr[ni].q[0] = *(const uint4*)&lB[cur][cB * LDSTRIDE + koff];
      bfr[ni].q[1] = *(const uint4*)&lB[cur][cB * LDSTRIDE + koff + 16];
    }
#pragma unroll
    for (int mi = 0; mi < 2; ++mi)
#pragma unroll
      for (int ni = 0; ni < 4; ++ni)
        acc[mi][ni] = __builtin_amdgcn_wmma_f32_16x16x32_bf16(
            false, af[mi].v, false, bfr[ni].v, (short)0, acc[mi][ni], false, false);

    async_fence_lds();   // next tile's async LDS writes complete before handoff
    __syncthreads();
  }

  // --- store D (bf16), optional bias2 add ---
  __bf16* outbase = hout + fb512 * NHW;
#pragma unroll
  for (int mi = 0; mi < 2; ++mi)
#pragma unroll
    for (int ni = 0; ni < 4; ++ni) {
      const int hw = hw0 + wn * 64 + ni * 16 + l15;
#pragma unroll
      for (int rr = 0; rr < 8; ++rr) {
        const int o = m0 + wm * 32 + mi * 16 + rr + lhi * 8;
        float v = acc[mi][ni][rr];
        if (STAGE == 2) v += bias2[b * NCH + o];
        outbase[(size_t)o * NHW + hw] = (__bf16)v;
      }
    }
}

// ---------------------------------------------------------------------------
// Per-(f,o) BN stats over (b,h,w): 4096 elems -> folded scale/shift
// ---------------------------------------------------------------------------
__global__ void k_stats(const __bf16* __restrict__ h, const float* __restrict__ g,
                        const float* __restrict__ bt, float* __restrict__ scale,
                        float* __restrict__ shift) {
  const int fo = blockIdx.x;
  const int f = fo >> 9, o = fo & 511;
  const int tid = threadIdx.x;
  const int bb = tid >> 4;
  const int hwl = tid & 15;
  const __bf16* p = h + ((size_t)(f * NBATCH + bb) * NCH + o) * NHW + hwl;
  float s = 0.f, sq = 0.f;
#pragma unroll
  for (int j = 0; j < 16; ++j) {
    const float v = (float)p[j * 16];
    s += v; sq += v * v;
  }
  __shared__ float rs[256], rq[256];
  rs[tid] = s; rq[tid] = sq;
  __syncthreads();
  for (int off = 128; off > 0; off >>= 1) {
    if (tid < off) { rs[tid] += rs[tid + off]; rq[tid] += rq[tid + off]; }
    __syncthreads();
  }
  if (tid == 0) {
    const float mean = rs[0] * (1.f / 4096.f);
    const float var  = rq[0] * (1.f / 4096.f) - mean * mean;
    const float rstd = rsqrtf(var + 1e-5f);
    const float sc = g[o] * rstd;
    scale[fo] = sc;
    shift[fo] = bt[o] - mean * sc;
  }
}

// ---------------------------------------------------------------------------
// BN2+leaky fused with 2x2 stride-2 single-channel conv -> out[f,b,64]
// ---------------------------------------------------------------------------
__global__ void k_conv3(const __bf16* __restrict__ h2, const float* __restrict__ scl,
                        const float* __restrict__ shf, const float* __restrict__ w3,
                        float* __restrict__ out) {
  const int fb = blockIdx.x;                  // f*16 + b
  const int f = fb >> 4;
  const int tid = threadIdx.x;
  const int s = tid & 63;
  const int y = s >> 3, x = s & 7;
  const int q = tid >> 6;                     // channel quarter
  const float* sc = scl + f * NCH;
  const float* sh = shf + f * NCH;
  const __bf16* base = h2 + (size_t)fb * NCH * NHW + (2 * y) * 16 + 2 * x;
  float acc = 0.f;
  for (int i = 0; i < 128; ++i) {
    const int c = q * 128 + i;
    const __bf16* p = base + (size_t)c * NHW;
    const float scc = sc[c], shc = sh[c];
    const float a00 = leaky((float)p[0]  * scc + shc);
    const float a01 = leaky((float)p[1]  * scc + shc);
    const float a10 = leaky((float)p[16] * scc + shc);
    const float a11 = leaky((float)p[17] * scc + shc);
    const float* w = w3 + c * 4;
    acc += a00 * w[0] + a01 * w[1] + a10 * w[2] + a11 * w[3];
  }
  __shared__ float red[256];
  red[tid] = acc;
  __syncthreads();
  if (tid < 64)
    out[(size_t)fb * 64 + s] = red[tid] + red[tid + 64] + red[tid + 128] + red[tid + 192];
}

// ---------------------------------------------------------------------------
extern "C" void kernel_launch(void* const* d_in, const int* in_sizes, int n_in,
                              void* d_out, int out_size, void* d_ws, size_t ws_size,
                              hipStream_t stream) {
  const float* cond    = (const float*)d_in[0];
  const float* xbar    = (const float*)d_in[1];
  const float* lin_w   = (const float*)d_in[2];
  const float* lin_b   = (const float*)d_in[3];
  const float* bn_s_g  = (const float*)d_in[4];
  const float* bn_s_b  = (const float*)d_in[5];
  const float* conv1_w = (const float*)d_in[6];
  const float* bn1_g   = (const float*)d_in[7];
  const float* bn1_b   = (const float*)d_in[8];
  const float* conv2_w = (const float*)d_in[9];
  const float* bn2_g   = (const float*)d_in[10];
  const float* bn2_b   = (const float*)d_in[11];
  const float* conv3_w = (const float*)d_in[12];
  float* out = (float*)d_out;

  char* ws = (char*)d_ws;
  float*  sent   = (float*)(ws + 0);                    //  16 KB
  float*  bias2  = (float*)(ws + 16384);                //  32 KB
  __bf16* wA1    = (__bf16*)(ws + 49152);               // 512 KB
  __bf16* wA2    = (__bf16*)(ws + 573440);              // 512 KB
  float*  scale1 = (float*)(ws + 1097728);              //  32 KB
  float*  shift1 = (float*)(ws + 1130496);
  float*  scale2 = (float*)(ws + 1163264);
  float*  shift2 = (float*)(ws + 1196032);
  __bf16* h1     = (__bf16*)(ws + 1228800);             //  64 MB
  __bf16* h2     = (__bf16*)(ws + 1228800 + (size_t)67108864);

  k_sent<<<1, 256, 0, stream>>>(cond, lin_w, lin_b, bn_s_g, bn_s_b, sent);
  k_cvt<<<1024, 256, 0, stream>>>(conv1_w, conv2_w, wA1, wA2);
  k_bias2<<<32, 256, 0, stream>>>(conv2_w, sent, bias2);

  k_gemm<1><<<2048, 256, 0, stream>>>(xbar, wA1, nullptr, nullptr, nullptr, nullptr, h1);
  k_stats<<<NFRAMES * NCH, 256, 0, stream>>>(h1, bn1_g, bn1_b, scale1, shift1);
  k_gemm<2><<<2048, 256, 0, stream>>>(xbar, wA2, h1, scale1, shift1, bias2, h2);
  k_stats<<<NFRAMES * NCH, 256, 0, stream>>>(h2, bn2_g, bn2_b, scale2, shift2);
  k_conv3<<<NFRAMES * NBATCH, 256, 0, stream>>>(h2, scale2, shift2, conv3_w, out);
}
